// feature_matching_KICS_2_85126251806892
// MI455X (gfx1250) — matrix operand order repaired and checked by reading, not compiled, not run on hardware
//
#include <hip/hip_runtime.h>

typedef __attribute__((ext_vector_type(16))) _Float16 v16h;
typedef __attribute__((ext_vector_type(8)))  _Float16 v8h;
typedef __attribute__((ext_vector_type(4)))  _Float16 v4h;
typedef __attribute__((ext_vector_type(8)))  float    v8f;
typedef __attribute__((ext_vector_type(4)))  float    v4f;

__device__ __forceinline__ float leaky(float x) { return x > 0.0f ? x : 0.1f * x; }

__device__ __forceinline__ v16h cat(v8h a, v8h b) {
    return __builtin_shufflevector(a, b, 0,1,2,3,4,5,6,7,8,9,10,11,12,13,14,15);
}

// ---------------------------------------------------------------------------
// Correlation: corr[b][di][dj][h][w] = sum_c rgb[b,c,h,w]*depth[b,c,h+di-10,w+dj-10]
// One block per (b,di,h). LDS rows transposed [w][c] -> 128-bit fragment loads.
// Two GEMMs per w-tile cover dj=0..19. Results staged in LDS (trash row for
// invalid dj -> branchless scatter) then dumped coalesced b128 to HBM.
// ---------------------------------------------------------------------------
__global__ __launch_bounds__(128) void corr_kernel(
    const float* __restrict__ rgb, const float* __restrict__ depth,
    float* __restrict__ corr)
{
    constexpr int C = 64, H = 64, W = 192, P = 20, WPD = 212; // 211 used
    constexpr int CP = 72;                                    // padded c-stride

    __shared__ alignas(32) _Float16 s_rgb[W   * CP];          // [w][c]
    __shared__ alignas(32) _Float16 s_dep[WPD * CP];          // [w+dj pad][c]
    __shared__ alignas(16) float    s_out[(P + 1) * W];       // row P = trash

    const int idx = blockIdx.x;
    const int b  = idx / (P * H);
    const int di = (idx / H) % P;
    const int h  = idx % H;
    const int tid = threadIdx.x;

    const int hh = h + di - 10;
    float* corr_row = corr + (((long)b * P + di) * P) * (H * (long)W) + (long)h * W;

    if (hh < 0 || hh >= H) {
        for (int i = tid; i < P * W / 4; i += 128) {
            int off = 4 * i;
            int dj = off / W, w4 = off % W;
            *(v4f*)&corr_row[(long)dj * H * W + w4] = (v4f){0.f, 0.f, 0.f, 0.f};
        }
        return;
    }

    const float* rgb_b = rgb   + ((long)b * C * H + h)  * W;
    const float* dep_b = depth + ((long)b * C * H + hh) * W;

    for (int i = tid; i < (C / 4) * W; i += 128) {
        int w = i % W, cg = i / W;
        v4h p = { (_Float16)rgb_b[(long)(4 * cg + 0) * H * W + w],
                  (_Float16)rgb_b[(long)(4 * cg + 1) * H * W + w],
                  (_Float16)rgb_b[(long)(4 * cg + 2) * H * W + w],
                  (_Float16)rgb_b[(long)(4 * cg + 3) * H * W + w] };
        *(v4h*)&s_rgb[w * CP + 4 * cg] = p;
    }
    for (int i = tid; i < (C / 4) * WPD; i += 128) {
        int wp = i % WPD, cg = i / WPD;
        int ws = wp - 10;
        bool ok = (wp < 211 && ws >= 0 && ws < W);
        v4h p = { (_Float16)(ok ? dep_b[(long)(4 * cg + 0) * H * W + ws] : 0.0f),
                  (_Float16)(ok ? dep_b[(long)(4 * cg + 1) * H * W + ws] : 0.0f),
                  (_Float16)(ok ? dep_b[(long)(4 * cg + 2) * H * W + ws] : 0.0f),
                  (_Float16)(ok ? dep_b[(long)(4 * cg + 3) * H * W + ws] : 0.0f) };
        *(v4h*)&s_dep[wp * CP + 4 * cg] = p;
    }
    __syncthreads();

    const int wave = tid >> 5, lane = tid & 31;
    const int lm = lane & 15, lh = lane >> 4;

    for (int t = wave; t < W / 16; t += 4) {
        const int w0 = t * 16;
        v8f acc1 = {}; v8f acc2 = {};
        #pragma unroll
        for (int kc = 0; kc < 2; ++kc) {        // K = 64 = 2 x 32
            const int kb = kc * 32;
            const _Float16* ar = &s_rgb[(w0 + lm) * CP + kb + 16 * lh];
            v16h a = cat(*(const v8h*)ar, *(const v8h*)(ar + 8));
            const _Float16* b1p = &s_dep[(w0 + lm) * CP + kb + 16 * lh];
            v16h b1 = cat(*(const v8h*)b1p, *(const v8h*)(b1p + 8));
            const _Float16* b2p = &s_dep[(w0 + 16 + lm) * CP + kb + 16 * lh];
            v16h b2 = cat(*(const v8h*)b2p, *(const v8h*)(b2p + 8));
            acc1 = __builtin_amdgcn_wmma_f32_16x16x32_f16(false, a, false, b1,
                                                          (short)0, acc1, false, false);
            acc2 = __builtin_amdgcn_wmma_f32_16x16x32_f16(false, a, false, b2,
                                                          (short)0, acc2, false, false);
        }
        // branchless scatter to LDS (invalid dj -> trash row P)
        #pragma unroll
        for (int r = 0; r < 8; ++r) {
            int m = r + 8 * lh;
            int w = w0 + m;
            int dj1 = lm - m;                        // D1 -> dj = n-m
            int row1 = (dj1 >= 0) ? dj1 : P;
            s_out[row1 * W + w] = acc1[r];
            int dj2 = 16 + lm - m;                   // D2 -> dj = 16+n-m
            int row2 = (dj2 < P) ? dj2 : P;
            s_out[row2 * W + w] = acc2[r];
        }
    }
    __syncthreads();

    // dense coalesced dump: 20 x 192 floats, 128-bit stores
    for (int i = tid; i < P * W / 4; i += 128) {
        int off = 4 * i;
        int dj = off / W, w4 = off % W;
        *(v4f*)&corr_row[(long)dj * H * W + w4] = *(const v4f*)&s_out[off];
    }
}

// ---------------------------------------------------------------------------
// Fused depthwise-separable block: out = leaky(pw(leaky(dw3x3x3(in))))
// Depthwise: register-blocked sliding window, B128 LDS reads.
// Pointwise: WMMA GEMM, A pre-swizzled fragment order, B stored [w][c].
// Output staged in LDS (aliases dead s_in) -> coalesced b128 stores.
// ---------------------------------------------------------------------------
template<int CIN, int COUT, int DIN, int HIN, int WIN, int S>
__global__ __launch_bounds__(128) void block_kernel(
    const float* __restrict__ in,
    const float* __restrict__ dw_w, const float* __restrict__ dw_b,
    const float* __restrict__ pw_w, const float* __restrict__ pw_b,
    float* __restrict__ out)
{
    constexpr int DOUT = (DIN - 1) / S + 1;
    constexpr int HOUT = (HIN - 1) / S + 1;
    constexpr int WOUT = (WIN - 1) / S + 1;
    constexpr int KP    = (CIN + 31) & ~31;
    constexpr int KC    = KP / 32;
    constexpr int MP    = (COUT + 15) & ~15;
    constexpr int MT    = MP / 16;
    constexpr int NT    = WOUT / 16;
    constexpr int KPp   = KP + 8;                  // bank-spread B row stride
    constexpr int WINP2 = ((WIN + 2 + 7) & ~7);    // halo row, 8-half aligned

    // s_in is dead after the depthwise phase; its storage is reused as the
    // fp32 output staging tile (MP*WOUT floats <= CIN*9*WINP2 halves' bytes).
    static_assert(MP * WOUT * 4 <= CIN * 9 * WINP2 * 2, "s_out overlay fits");

    __shared__ alignas(32) _Float16 s_in[CIN * 9 * WINP2]; // [c][kd*3+kh][wi]
    __shared__ alignas(32) _Float16 s_dw[WOUT * KPp];      // B operand [w][c]
    __shared__ alignas(32) _Float16 s_A[MT * KC * 32 * 16];// fragment-order pw weights
    __shared__ float s_dww[CIN * 27];
    __shared__ float s_dwb[CIN];
    __shared__ float s_pwb[MP];                            // zero-padded bias

    const int idx = blockIdx.x;
    const int b  = idx / (DOUT * HOUT);
    const int rr = idx % (DOUT * HOUT);
    const int od = rr / HOUT;
    const int oh = rr % HOUT;
    const int tid = threadIdx.x;

    for (int i = tid; i < CIN * 27; i += 128) s_dww[i] = dw_w[i];
    for (int i = tid; i < CIN;      i += 128) s_dwb[i] = dw_b[i];
    for (int i = tid; i < MP;       i += 128) s_pwb[i] = (i < COUT) ? pw_b[i] : 0.0f;
    for (int i = tid; i < MT * KC * 32 * 16; i += 128) {
        int j    = i & 15;
        int lane = (i >> 4) & 31;
        int kc   = (i >> 9) % KC;
        int mt   = (i >> 9) / KC;
        int m = mt * 16 + (lane & 15);
        int k = kc * 32 + 16 * (lane >> 4) + j;
        s_A[i] = (_Float16)((m < COUT && k < CIN) ? pw_w[m * CIN + k] : 0.0f);
    }
    for (int i = tid; i < CIN * 9 * (WINP2 / 4); i += 128) {
        int wq  = i % (WINP2 / 4);
        int t   = (i / (WINP2 / 4)) % 9;
        int c   = i / (9 * (WINP2 / 4));
        int kd  = t / 3, kh = t % 3;
        int d_in = od * S + kd - 1;
        int h_in = oh * S + kh - 1;
        bool rowok = (d_in >= 0 && d_in < DIN && h_in >= 0 && h_in < HIN);
        const float* src = in + (((long)b * CIN + c) * DIN + d_in) * ((long)HIN * WIN)
                              + (long)h_in * WIN;
        v4h p;
        #pragma unroll
        for (int q = 0; q < 4; ++q) {
            int w_in = 4 * wq + q - 1;
            p[q] = (_Float16)((rowok && w_in >= 0 && w_in < WIN) ? src[w_in] : 0.0f);
        }
        *(v4h*)&s_in[(c * 9 + t) * WINP2 + 4 * wq] = p;
    }
    __syncthreads();

    // depthwise 3x3x3 + leaky: task = 8 consecutive w outputs of one channel
    constexpr int NC8 = WOUT / 8;
    for (int i = tid; i < KP * NC8; i += 128) {
        int c  = i / NC8;
        int w0 = (i % NC8) * 8;
        if (c >= CIN) {
            #pragma unroll
            for (int ow = 0; ow < 8; ++ow) s_dw[(w0 + ow) * KPp + c] = (_Float16)0.0f;
        } else {
            float acc[8];
            #pragma unroll
            for (int ow = 0; ow < 8; ++ow) acc[ow] = s_dwb[c];
            const _Float16* base = &s_in[c * 9 * WINP2];
            const float*    wt   = &s_dww[c * 27];
            #pragma unroll
            for (int t = 0; t < 9; ++t) {
                const _Float16* row = base + t * WINP2 + w0 * S;
                v8h r0 = *(const v8h*)(row);
                v8h r1 = *(const v8h*)(row + 8);
                v8h r2;
                if constexpr (S == 2) r2 = *(const v8h*)(row + 16);
                else                  r2 = r1;
                float wt0 = wt[t * 3 + 0], wt1 = wt[t * 3 + 1], wt2 = wt[t * 3 + 2];
                #define EL(ii) ((ii) < 8 ? r0[(ii)] : ((ii) < 16 ? r1[(ii) - 8] : r2[(ii) - 16]))
                #pragma unroll
                for (int ow = 0; ow < 8; ++ow) {
                    acc[ow] += (float)EL(ow * S + 0) * wt0
                             + (float)EL(ow * S + 1) * wt1
                             + (float)EL(ow * S + 2) * wt2;
                }
                #undef EL
            }
            #pragma unroll
            for (int ow = 0; ow < 8; ++ow)
                s_dw[(w0 + ow) * KPp + c] = (_Float16)leaky(acc[ow]);
        }
    }
    __syncthreads();

    // pointwise GEMM via WMMA -> LDS staging tile (no predicated stores)
    float* s_out = reinterpret_cast<float*>(s_in);
    const int wave = tid >> 5, lane = tid & 31;
    const int lm = lane & 15, lh = lane >> 4;

    for (int task = wave; task < NT * MT; task += 4) {
        const int mt = task / NT;
        const int w0 = (task % NT) * 16;
        v8f acc = {};
        #pragma unroll
        for (int kc = 0; kc < KC; ++kc) {
            const v8h* ap = (const v8h*)&s_A[((mt * KC + kc) * 32 + lane) * 16];
            v16h a = cat(ap[0], ap[1]);
            const _Float16* bp = &s_dw[(w0 + lm) * KPp + kc * 32 + 16 * lh];
            v16h bm = cat(*(const v8h*)bp, *(const v8h*)(bp + 8));
            acc = __builtin_amdgcn_wmma_f32_16x16x32_f16(false, a, false, bm,
                                                         (short)0, acc, false, false);
        }
        #pragma unroll
        for (int r = 0; r < 8; ++r) {
            int m = mt * 16 + r + 8 * lh;
            s_out[m * WOUT + w0 + lm] = leaky(acc[r] + s_pwb[m]);
        }
    }
    __syncthreads();

    // dense coalesced write-out of the valid COUT x WOUT region
    for (int i = tid; i < COUT * WOUT / 4; i += 128) {
        int off = 4 * i;
        int m = off / WOUT, w4 = off % WOUT;
        *(v4f*)&out[(((long)b * COUT + m) * DOUT + od) * ((long)HOUT * WOUT)
                    + (long)oh * WOUT + w4] = *(const v4f*)&s_out[off];
    }
}

// ---------------------------------------------------------------------------
extern "C" void kernel_launch(void* const* d_in, const int* in_sizes, int n_in,
                              void* d_out, int out_size, void* d_ws, size_t ws_size,
                              hipStream_t stream) {
    (void)in_sizes; (void)n_in; (void)out_size; (void)ws_size;

    const float* rgb   = (const float*)d_in[0];
    const float* depth = (const float*)d_in[1];
    const float* dw1w = (const float*)d_in[2];  const float* dw1b = (const float*)d_in[3];
    const float* pw1w = (const float*)d_in[4];  const float* pw1b = (const float*)d_in[5];
    const float* dw2w = (const float*)d_in[6];  const float* dw2b = (const float*)d_in[7];
    const float* pw2w = (const float*)d_in[8];  const float* pw2b = (const float*)d_in[9];
    const float* dw3w = (const float*)d_in[10]; const float* dw3b = (const float*)d_in[11];
    const float* pw3w = (const float*)d_in[12]; const float* pw3b = (const float*)d_in[13];
    const float* dw4w = (const float*)d_in[14]; const float* dw4b = (const float*)d_in[15];
    const float* pw4w = (const float*)d_in[16]; const float* pw4b = (const float*)d_in[17];

    // ws ping-pong: segment A @0, segment B @157286400 (each 157,286,400 B)
    char*  ws   = (char*)d_ws;
    float* corr = (float*)ws;                   // (8,20,20,64,192) f32
    float* buf1 = (float*)(ws + 157286400);     // block1 out
    float* buf2 = (float*)ws;                   // block2 out (8,40,10,32,96)
    float* buf3 = (float*)(ws + 157286400);     // block3 out

    corr_kernel<<<8 * 20 * 64, 128, 0, stream>>>(rgb, depth, corr);

    block_kernel<20, 20, 20, 64, 192, 1>
        <<<8 * 20 * 64, 128, 0, stream>>>(corr, dw1w, dw1b, pw1w, pw1b, buf1);
    block_kernel<20, 40, 20, 64, 192, 2>
        <<<8 * 10 * 32, 128, 0, stream>>>(buf1, dw2w, dw2b, pw2w, pw2b, buf2);
    block_kernel<40, 40, 10, 32, 96, 1>
        <<<8 * 10 * 32, 128, 0, stream>>>(buf2, dw3w, dw3b, pw3w, pw3b, buf3);
    block_kernel<40, 80, 10, 32, 96, 2>
        <<<8 * 5 * 16, 128, 0, stream>>>(buf3, dw4w, dw4b, pw4w, pw4b, (float*)d_out);
}
